// SS2DCore_87445534146662
// MI455X (gfx1250) — compile-verified, matrix operand order checked by reading, not run on hardware
//
#include <hip/hip_runtime.h>
#include <hip/hip_bf16.h>
#include <stdint.h>

#define DEVI __device__ __forceinline__

typedef __attribute__((ext_vector_type(16))) __bf16 v16bf;
typedef __attribute__((ext_vector_type(8)))  float  v8f;

union U16x16 { uint4 q[2]; v16bf v; };

DEVI uint16_t f2bf(float f) {
  uint32_t u = __float_as_uint(f);
  uint32_t r = u + 0x7FFFu + ((u >> 16) & 1u);   // round-to-nearest-even
  return (uint16_t)(r >> 16);
}
DEVI float bf2f(uint16_t h) { return __uint_as_float(((uint32_t)h) << 16); }

// ---------------- problem constants ----------------
constexpr int BATCH = 2, HH = 128, WWID = 128;
constexpr int LSEQ = HH * WWID;          // 16384
constexpr int TTOK = BATCH * LSEQ;       // 32768
constexpr int CIN = 96, NLAT = 256;
constexpr int CHUNK = 256, NCHUNK = LSEQ / CHUNK;   // 64

// ---------------- workspace layout (bytes) ----------------
constexpr size_t SZT = (size_t)TTOK;
constexpr size_t OFF_P     = 0;                                  // proj fp32 [T,1024]
constexpr size_t OFF_ABAR  = OFF_P     + SZT * 1024 * 4;         // fp32 [T,256]
constexpr size_t OFF_BU    = OFF_ABAR  + SZT * 256 * 4;          // fp32 [T,256]
constexpr size_t OFF_CMB   = OFF_BU    + SZT * 256 * 4;          // bf16 [T,256]
constexpr size_t OFF_XTOK  = OFF_CMB   + SZT * 256 * 2;          // bf16 [T,96]
constexpr size_t OFF_WPROJ = OFF_XTOK  + SZT * 96 * 2;           // bf16 [1024,96]
constexpr size_t OFF_BPROJ = OFF_WPROJ + (size_t)1024 * 96 * 2;  // fp32 [1024]
constexpr size_t OFF_YWB   = OFF_BPROJ + 1024 * 4;               // bf16 [256,256]
constexpr size_t OFF_WMG   = OFF_YWB   + (size_t)256 * 256 * 2;  // bf16 [512,1024]
constexpr size_t OFF_BMG   = OFF_WMG   + (size_t)512 * 1024 * 2; // fp32 [512]
constexpr size_t OFF_WOW   = OFF_BMG   + 512 * 4;                // bf16 [96,256]
constexpr size_t OFF_CHA   = OFF_WOW   + (size_t)96 * 256 * 2;   // fp32 [512*256]
constexpr size_t OFF_CHB   = OFF_CHA   + (size_t)512 * 256 * 4;
constexpr size_t OFF_CARRY = OFF_CHB   + (size_t)512 * 256 * 4;
// aliases (regions freed after earlier stages)
constexpr size_t OFF_YDIR  = OFF_P;                              // bf16 [4][T][256] (64MB)
constexpr size_t OFF_G     = OFF_P + (size_t)4 * SZT * 256 * 2;  // bf16 [T][1024]   (64MB)
constexpr size_t OFF_MG    = OFF_ABAR;                           // fp32 [T][512]    (64MB over ABAR+BU)
constexpr size_t OFF_Z     = OFF_CMB;                            // bf16 [T][256]

// ---------------- small prep kernels ----------------
__global__ void cvt_bf16_k(uint16_t* __restrict__ dst, const float* __restrict__ src, int n) {
  int i = blockIdx.x * 256 + threadIdx.x;
  if (i < n) dst[i] = f2bf(src[i]);
}
__global__ void copy_f32_k(float* __restrict__ dst, const float* __restrict__ src, int n) {
  int i = blockIdx.x * 256 + threadIdx.x;
  if (i < n) dst[i] = src[i];
}
// x (B,C,H,W) fp32 -> xtok bf16 [T][96] (token-major)
__global__ void xtok_k(const float* __restrict__ x, uint16_t* __restrict__ xt) {
  int i = blockIdx.x * 256 + threadIdx.x;
  if (i >= TTOK * CIN) return;
  int c = i % CIN;
  int t = i / CIN;
  int b = t >> 14, l = t & (LSEQ - 1);
  xt[i] = f2bf(x[((size_t)(b * CIN + c) << 14) + l]);
}

// ------- WMMA bf16 GEMM, register-blocked MT x NT tiles per wave -------
// C[M,N] = A[M,K] * W[N,K]^T + bias
// MODE 0: fp32 row-major out (ldc). MODE 1: bf16 row-major out (ldc).
// MODE 2: final scatter to (B, 96, H, W) fp32.
template<int MODE, int MT, int NT>
__global__ __launch_bounds__(256) void gemm_bf16(
    const uint16_t* __restrict__ A, const uint16_t* __restrict__ W,
    const float* __restrict__ bias, float* __restrict__ outF,
    uint16_t* __restrict__ outH, int M, int N, int K, int ldc)
{
  int wave    = (blockIdx.x * 256 + threadIdx.x) >> 5;
  int groupsN = N / (16 * NT);
  int gm      = wave / groupsN;
  int gn      = wave - gm * groupsN;
  if (gm >= M / (16 * MT)) return;         // wave-uniform: EXEC stays all-ones
  int lane = threadIdx.x & 31;
  int half = lane >> 4;
  int r    = lane & 15;

  const uint16_t* arow[MT];
  const uint16_t* brow[NT];
#pragma unroll
  for (int mt = 0; mt < MT; mt++)
    arow[mt] = A + (size_t)((gm * MT + mt) * 16 + r) * K + half * 8;
#pragma unroll
  for (int nt = 0; nt < NT; nt++)
    brow[nt] = W + (size_t)((gn * NT + nt) * 16 + r) * K + half * 8;

  v8f acc[MT][NT];
#pragma unroll
  for (int mt = 0; mt < MT; mt++)
#pragma unroll
    for (int nt = 0; nt < NT; nt++)
      acc[mt][nt] = (v8f){0.f, 0.f, 0.f, 0.f, 0.f, 0.f, 0.f, 0.f};

  for (int k0 = 0; k0 < K; k0 += 32) {
    U16x16 av[MT], bv[NT];
#pragma unroll
    for (int mt = 0; mt < MT; mt++) {
      av[mt].q[0] = *(const uint4*)(arow[mt] + k0);
      av[mt].q[1] = *(const uint4*)(arow[mt] + k0 + 16);
    }
#pragma unroll
    for (int nt = 0; nt < NT; nt++) {
      bv[nt].q[0] = *(const uint4*)(brow[nt] + k0);
      bv[nt].q[1] = *(const uint4*)(brow[nt] + k0 + 16);
    }
    __builtin_prefetch((const void*)(arow[0] + k0 + 128), 0, 1);  // global_prefetch_b8
#pragma unroll
    for (int mt = 0; mt < MT; mt++)
#pragma unroll
      for (int nt = 0; nt < NT; nt++)
        acc[mt][nt] = __builtin_amdgcn_wmma_f32_16x16x32_bf16(
            false, av[mt].v, false, bv[nt].v, (short)0, acc[mt][nt], false, false);
  }

#pragma unroll
  for (int nt = 0; nt < NT; nt++) {
    int n = (gn * NT + nt) * 16 + r;
    float bb = bias ? bias[n] : 0.f;
#pragma unroll
    for (int mt = 0; mt < MT; mt++) {
#pragma unroll
      for (int i = 0; i < 8; i++) {
        int m = (gm * MT + mt) * 16 + i + half * 8;
        float val = acc[mt][nt][i] + bb;
        if (MODE == 0)       outF[(size_t)m * ldc + n] = val;
        else if (MODE == 1)  outH[(size_t)m * ldc + n] = f2bf(val);
        else {
          int b = m >> 14, p = m & (LSEQ - 1);
          outF[((size_t)(b * 96 + n) << 14) + p] = val;
        }
      }
    }
  }
}

// ---------------- elementwise: A_bar, B_bar*u, Cm->bf16 ----------------
__global__ void stage2_k(const float* __restrict__ P, float* __restrict__ Ab,
                         float* __restrict__ Bu, uint16_t* __restrict__ Cmb) {
  size_t i = (size_t)blockIdx.x * 256 + threadIdx.x;   // T*256 threads
  int n = (int)(i & 255);
  size_t t = i >> 8;
  const float* row = P + t * 1024;
  float u  = row[n];
  float Bm = row[256 + n];
  float Cm = row[512 + n];
  float de = row[768 + n];
  float sp = fmaxf(de, 0.f) + log1pf(expf(-fabsf(de)));   // stable softplus
  float a  = expf(-sp);
  Ab[i]  = a;
  Bu[i]  = (1.f - a) * Bm * u;
  Cmb[i] = f2bf(Cm);
}

// direction-position p -> token index (b*L + l) in s1 memory order
DEVI size_t tokidx(int d, int b, int pos) {
  int pp = (d >= 2) ? (LSEQ - 1 - pos) : pos;
  int l;
  if (d == 1 || d == 3) {                 // s2/s4: pos = w*H + h ; l = h*W + w
    int h = pp & (HH - 1);
    int w = pp >> 7;
    l = h * WWID + w;
  } else l = pp;
  return (size_t)(b * LSEQ + l);
}

// ---------------- 3-phase chunked scan ----------------
// blockIdx.x in [0,512): d = bx>>7, b = (bx>>6)&1, chunk = bx&63; 256 threads = channels
__global__ void scan_phase1_k(const float* __restrict__ Ab, const float* __restrict__ Bu,
                              float* __restrict__ chA, float* __restrict__ chB) {
  int bx = blockIdx.x;
  int d = bx >> 7, b = (bx >> 6) & 1, chunk = bx & 63;
  int n = threadIdx.x;
  float Ar = 1.f, Br = 0.f;
  for (int i = 0; i < CHUNK; i++) {
    size_t t = tokidx(d, b, chunk * CHUNK + i);
    float a = Ab[t * 256 + n];
    float v = Bu[t * 256 + n];
    Ar *= a;
    Br = a * Br + v;
  }
  chA[(size_t)bx * 256 + n] = Ar;
  chB[(size_t)bx * 256 + n] = Br;
}
__global__ void scan_phase2_k(const float* __restrict__ chA, const float* __restrict__ chB,
                              float* __restrict__ carry) {
  int s = blockIdx.x;          // d*2 + b in [0,8)
  int n = threadIdx.x;
  float Br = 0.f;
  for (int c = 0; c < NCHUNK; c++) {
    size_t idx = ((size_t)(s * NCHUNK + c)) * 256 + n;
    carry[idx] = Br;                    // incoming h for this chunk
    Br = chA[idx] * Br + chB[idx];
  }
}
__global__ void scan_phase3_k(const float* __restrict__ Ab, const float* __restrict__ Bu,
                              const uint16_t* __restrict__ Cmb,
                              const float* __restrict__ carry,
                              uint16_t* __restrict__ ydir) {
  int bx = blockIdx.x;
  int d = bx >> 7, b = (bx >> 6) & 1, chunk = bx & 63;
  int n = threadIdx.x;
  float h = carry[(size_t)bx * 256 + n];
  for (int i = 0; i < CHUNK; i++) {
    int pos = chunk * CHUNK + i;
    size_t t = tokidx(d, b, pos);
    float a = Ab[t * 256 + n];
    float v = Bu[t * 256 + n];
    h = a * h + v;
    float y = bf2f(Cmb[t * 256 + n]) * h;
    ydir[(((size_t)d * TTOK) + (size_t)b * LSEQ + pos) * 256 + n] = f2bf(y);
  }
}

// ---------------- z = merged * sigmoid(gate) ----------------
__global__ void gatefuse_k(const float* __restrict__ MG, uint16_t* __restrict__ Z) {
  size_t i = (size_t)blockIdx.x * 256 + threadIdx.x;   // T*256
  int n = (int)(i & 255);
  size_t t = i >> 8;
  float m = MG[t * 512 + n];
  float g = MG[t * 512 + 256 + n];
  Z[i] = f2bf(m * (1.f / (1.f + expf(-g))));
}

// ---------------- host side ----------------
template<int MODE, int MT, int NT>
static void launch_gemm(const uint16_t* A, const uint16_t* W, const float* bias,
                        float* outF, uint16_t* outH, int M, int N, int K, int ldc,
                        hipStream_t s) {
  int waves = (M / (16 * MT)) * (N / (16 * NT));
  int blocks = (waves + 7) / 8;
  gemm_bf16<MODE, MT, NT><<<blocks, 256, 0, s>>>(A, W, bias, outF, outH, M, N, K, ldc);
}

extern "C" void kernel_launch(void* const* d_in, const int* in_sizes, int n_in,
                              void* d_out, int out_size, void* d_ws, size_t ws_size,
                              hipStream_t stream) {
  (void)in_sizes; (void)n_in; (void)out_size; (void)ws_size;
  uint8_t* ws = (uint8_t*)d_ws;
  const float* x   = (const float*)d_in[0];
  const float* xw  = (const float*)d_in[1];
  const float* xb  = (const float*)d_in[2];
  const float* bcw = (const float*)d_in[3];
  const float* bcb = (const float*)d_in[4];
  const float* dw  = (const float*)d_in[5];
  const float* db  = (const float*)d_in[6];
  const float* yw  = (const float*)d_in[7];
  const float* yb  = (const float*)d_in[8];
  const float* mw  = (const float*)d_in[9];
  const float* mb  = (const float*)d_in[10];
  const float* gw  = (const float*)d_in[11];
  const float* gb  = (const float*)d_in[12];
  const float* ow  = (const float*)d_in[13];
  const float* ob  = (const float*)d_in[14];

  auto CV = [&](size_t off, const float* s, int n) {
    cvt_bf16_k<<<(n + 255) / 256, 256, 0, stream>>>((uint16_t*)(ws + off), s, n);
  };
  auto CP = [&](size_t off, const float* s, int n) {
    copy_f32_k<<<(n + 255) / 256, 256, 0, stream>>>((float*)(ws + off), s, n);
  };

  // weight prep: Wproj = [xw; bcw; dw] (all already [Nout,K] row-major)
  CV(OFF_WPROJ,                        xw,  256 * 96);
  CV(OFF_WPROJ + (size_t)256 * 96 * 2, bcw, 512 * 96);
  CV(OFF_WPROJ + (size_t)768 * 96 * 2, dw,  256 * 96);
  CP(OFF_BPROJ,            xb,  256);
  CP(OFF_BPROJ + 256 * 4,  bcb, 512);
  CP(OFF_BPROJ + 768 * 4,  db,  256);
  CV(OFF_YWB, yw, 256 * 256);
  CV(OFF_WMG,                          mw, 256 * 1024);
  CV(OFF_WMG + (size_t)256 * 1024 * 2, gw, 256 * 1024);
  CP(OFF_BMG,           mb, 256);
  CP(OFF_BMG + 256 * 4, gb, 256);
  CV(OFF_WOW, ow, 96 * 256);

  // tokenize input
  xtok_k<<<(TTOK * CIN + 255) / 256, 256, 0, stream>>>(x, (uint16_t*)(ws + OFF_XTOK));

  // stage 1: proj[T,1024] = xtok @ Wproj^T + bias
  launch_gemm<0, 2, 4>((const uint16_t*)(ws + OFF_XTOK), (const uint16_t*)(ws + OFF_WPROJ),
                       (const float*)(ws + OFF_BPROJ), (float*)(ws + OFF_P), nullptr,
                       TTOK, 1024, 96, 1024, stream);

  // stage 2: A_bar, B_bar*u, Cm
  stage2_k<<<TTOK, 256, 0, stream>>>((const float*)(ws + OFF_P),
                                     (float*)(ws + OFF_ABAR),
                                     (float*)(ws + OFF_BU),
                                     (uint16_t*)(ws + OFF_CMB));

  // stage 3: chunked scans, all 4 directions
  scan_phase1_k<<<512, 256, 0, stream>>>((const float*)(ws + OFF_ABAR),
                                         (const float*)(ws + OFF_BU),
                                         (float*)(ws + OFF_CHA),
                                         (float*)(ws + OFF_CHB));
  scan_phase2_k<<<8, 256, 0, stream>>>((const float*)(ws + OFF_CHA),
                                       (const float*)(ws + OFF_CHB),
                                       (float*)(ws + OFF_CARRY));
  scan_phase3_k<<<512, 256, 0, stream>>>((const float*)(ws + OFF_ABAR),
                                         (const float*)(ws + OFF_BU),
                                         (const uint16_t*)(ws + OFF_CMB),
                                         (const float*)(ws + OFF_CARRY),
                                         (uint16_t*)(ws + OFF_YDIR));

  // stage 4: per direction, G[:, d*256:(d+1)*256] = y_d @ yw^T + yb  (bf16)
  for (int d = 0; d < 4; d++) {
    launch_gemm<1, 2, 4>((const uint16_t*)(ws + OFF_YDIR) + (size_t)d * TTOK * 256,
                         (const uint16_t*)(ws + OFF_YWB), yb, nullptr,
                         (uint16_t*)(ws + OFF_G) + (size_t)d * 256,
                         TTOK, 256, 256, 1024, stream);
  }

  // stage 5: MG[T,512] = G @ [mw;gw]^T + [mb;gb]
  launch_gemm<0, 2, 4>((const uint16_t*)(ws + OFF_G), (const uint16_t*)(ws + OFF_WMG),
                       (const float*)(ws + OFF_BMG), (float*)(ws + OFF_MG), nullptr,
                       TTOK, 512, 1024, 512, stream);

  // z = merged * sigmoid(gate)
  gatefuse_k<<<TTOK, 256, 0, stream>>>((const float*)(ws + OFF_MG),
                                       (uint16_t*)(ws + OFF_Z));

  // stage 6: out(B,96,H,W) = z @ ow^T + ob, scatter-transposed (N=96 -> NT=3)
  launch_gemm<2, 2, 3>((const uint16_t*)(ws + OFF_Z), (const uint16_t*)(ws + OFF_WOW),
                       ob, (float*)d_out, nullptr,
                       TTOK, 96, 256, 0, stream);
}
